// Attention_28544352649762
// MI455X (gfx1250) — compile-verified, matrix operand order checked by reading, not run on hardware
//
#include <hip/hip_runtime.h>

// ---------------- problem constants ----------------
#define B_ 2
#define N_ 2048
#define C_ 1024
#define H_ 16
#define D_ 64

typedef __bf16 bf16_t;
typedef bf16_t v8bf  __attribute__((ext_vector_type(8)));
typedef bf16_t v16bf __attribute__((ext_vector_type(16)));
typedef float  v8f   __attribute__((ext_vector_type(8)));
typedef unsigned int u32x4 __attribute__((ext_vector_type(4)));
typedef int          i32x4 __attribute__((ext_vector_type(4)));
typedef int          i32x8 __attribute__((ext_vector_type(8)));

// Build a 16-element A/B WMMA fragment from two contiguous 16-byte chunks.
__device__ __forceinline__ v16bf load_frag(const bf16_t* p0, const bf16_t* p1) {
    v8bf a = *(const v8bf*)p0;
    v8bf b = *(const v8bf*)p1;
    v16bf r;
#pragma unroll
    for (int i = 0; i < 8; ++i) { r[i] = a[i]; r[i + 8] = b[i]; }
    return r;
}

__device__ __forceinline__ v8f wmma_bf16(v16bf a, v16bf b, v8f c) {
    return __builtin_amdgcn_wmma_f32_16x16x32_bf16(
        /*neg_a=*/false, a, /*neg_b=*/false, b,
        /*c_mod=*/(short)0, c, /*reuse_a=*/false, /*reuse_b=*/false);
}

// ---- TDM: load a 512-row x 32-col bf16 tile (rows stride 1024 elems) to LDS ----
// D# group0: count=1, lds_addr, 57-bit global tile address, type=2 ("image").
// D# group1: data_size=2B, tensor_dim0/1=1024, tile_dim0=32, tile_dim1=512,
//            tensor_dim0_stride=1024 elements.
__device__ __forceinline__ void tdm_load_panel(const bf16_t* gptr, unsigned ldsOff) {
    unsigned long long ga = (unsigned long long)(uintptr_t)gptr;
    u32x4 g0;
    g0[0] = 1u;                                   // count=1, user mode
    g0[1] = ldsOff;                               // lds_addr (bytes)
    g0[2] = (unsigned)ga;                         // global_addr[31:0]
    g0[3] = (unsigned)((ga >> 32) & 0x01FFFFFFu)  // global_addr[56:32]
          | (2u << 30);                           // type=2
    i32x8 g1;
    g1[0] = (int)(1u << 16);                      // data_size=1 (2 bytes)
    g1[1] = (int)(1024u << 16);                   // tensor_dim0[15:0] @ bits 63:48
    g1[2] = (int)(1024u << 16);                   // tensor_dim1[15:0] @ bits 95:80
    g1[3] = (int)(32u << 16);                     // tile_dim0=32 @ bits 127:112
    g1[4] = 512;                                  // tile_dim1=512 @ bits 143:128
    g1[5] = 1024;                                 // tensor_dim0_stride[31:0]
    g1[6] = 0;
    g1[7] = 0;
    i32x4 z4; z4[0] = z4[1] = z4[2] = z4[3] = 0;
#if defined(__clang_major__) && (__clang_major__ >= 23)
    i32x8 z8; z8[0]=z8[1]=z8[2]=z8[3]=z8[4]=z8[5]=z8[6]=z8[7]=0;
    __builtin_amdgcn_tensor_load_to_lds(g0, g1, z4, z4, z8, 0);
#else
    __builtin_amdgcn_tensor_load_to_lds(g0, g1, z4, z4, 0);
#endif
}

// ---------------- elementwise conversion kernels ----------------
__global__ void cvt_f32_bf16(const float* __restrict__ in, bf16_t* __restrict__ out, int n) {
    int i = blockIdx.x * blockDim.x + threadIdx.x;
    if (i < n) out[i] = (bf16_t)in[i];
}

// W[k][n] (KxN row-major f32) -> Wt[n][k] (NxK row-major bf16)
__global__ void cvt_wT_bf16(const float* __restrict__ W, bf16_t* __restrict__ Wt) {
    int i = blockIdx.x * blockDim.x + threadIdx.x;   // over C_*C_
    if (i >= C_ * C_) return;
    int k = i & (C_ - 1);
    int n = i >> 10;
    Wt[(size_t)n * C_ + k] = (bf16_t)W[(size_t)k * C_ + n];
}

// In-place RoPE (fp32 math on bf16 storage), optional scaling folded in.
__global__ void rope_bf16(bf16_t* __restrict__ buf, const float* __restrict__ cosb,
                          const float* __restrict__ sinb, float scale) {
    int i = blockIdx.x * blockDim.x + threadIdx.x;   // B_*N_*H_*32
    if (i >= B_ * N_ * H_ * 32) return;
    int d  = i & 31;
    int h  = (i >> 5) & (H_ - 1);
    int bn = i >> 9;                                  // b*N_ + n
    size_t base = (size_t)bn * C_ + h * D_;
    float x1 = (float)buf[base + d];
    float x2 = (float)buf[base + d + 32];
    float c  = cosb[(size_t)bn * D_ + d];
    float s  = sinb[(size_t)bn * D_ + d];
    buf[base + d]      = (bf16_t)((x1 * c - x2 * s) * scale);
    buf[base + d + 32] = (bf16_t)((x2 * c + x1 * s) * scale);
}

// V [b][n][h*64+d] -> Vt [b][h][d][n]
__global__ void vT_bf16(const bf16_t* __restrict__ V, bf16_t* __restrict__ Vt) {
    int i = blockIdx.x * blockDim.x + threadIdx.x;   // B_*H_*D_*N_
    if (i >= B_ * H_ * D_ * N_) return;
    int n = i & (N_ - 1);
    int d = (i >> 11) & (D_ - 1);
    int h = (i >> 17) & (H_ - 1);
    int b = i >> 21;
    Vt[i] = V[((size_t)(b * N_ + n)) * C_ + h * D_ + d];
}

// ---------------- TDM-fed WMMA GEMM ----------------
// C[4096x1024] = A[4096x1024] * Bt[1024x1024]^T, bf16 in, fp32 acc.
// Block = 8 waves = one 16x512 C strip. B panel (512 rows x 32 k) staged in LDS
// by the Tensor Data Mover, double buffered, one outstanding TDM op.
__global__ void gemm_bf16_tdm(const bf16_t* __restrict__ A, const bf16_t* __restrict__ Bt,
                              float* __restrict__ Cf, bf16_t* __restrict__ Cb) {
    __shared__ bf16_t tileB[2][512 * 32];             // 2 x 32 KB
    const int w    = threadIdx.x >> 5;                // wave in block: 0..7
    const int lane = threadIdx.x & 31;
    const int ln = lane & 15, lh = lane >> 4;
    const int tm     = blockIdx.x >> 1;               // 16-row M tile
    const int tnBase = (blockIdx.x & 1) * 8;          // first 64-col N tile of strip

    const bf16_t* arow = A  + (size_t)(tm * 16 + ln) * C_;
    const bf16_t* gB   = Bt + (size_t)(tnBase * 64) * C_;   // panel row base
    const unsigned lds0 = (unsigned)(uintptr_t)&tileB[0][0];
    const unsigned lds1 = (unsigned)(uintptr_t)&tileB[1][0];

    v8f acc[4] = {};

    if (threadIdx.x < 32) tdm_load_panel(gB, lds0);   // prologue: panel for kb=0

    for (int i = 0; i < 32; ++i) {
        const int kb = i * 32;
        if (threadIdx.x < 32) __builtin_amdgcn_s_wait_tensorcnt((short)0);
        __syncthreads();                              // publish panel i to all waves
        if ((threadIdx.x < 32) && (i + 1 < 32))       // prefetch panel i+1
            tdm_load_panel(gB + (kb + 32), (i & 1) ? lds0 : lds1);

        v16bf af = load_frag(arow + kb + lh * 8, arow + kb + 16 + lh * 8);
        const bf16_t* lb = &tileB[i & 1][0];
#pragma unroll
        for (int j = 0; j < 4; ++j) {
            const bf16_t* br = lb + (size_t)(w * 64 + j * 16 + ln) * 32;
            v16bf bfr = load_frag(br + lh * 8, br + 16 + lh * 8);
            acc[j] = wmma_bf16(af, bfr, acc[j]);
        }
        __syncthreads();                              // panel i consumed -> reusable
    }

#pragma unroll
    for (int j = 0; j < 4; ++j) {
        int col = (tnBase + w) * 64 + j * 16 + ln;
#pragma unroll
        for (int r = 0; r < 8; ++r) {
            int row = tm * 16 + r + 8 * lh;
            if (Cf) Cf[(size_t)row * C_ + col] = acc[j][r];
            else    Cb[(size_t)row * C_ + col] = (bf16_t)acc[j][r];
        }
    }
}

// ---------------- Flash attention ----------------
// One wave per (b, h, 16-query block). 32-key blocks, online softmax fp32.
__global__ void flash_attn(const bf16_t* __restrict__ Q, const bf16_t* __restrict__ Kb,
                           const bf16_t* __restrict__ Vt, bf16_t* __restrict__ Ob) {
    __shared__ bf16_t lds[4][16 * 32];        // 1KB P-tile per wave
    int wid  = (blockIdx.x * blockDim.x + threadIdx.x) >> 5;
    int wloc = (threadIdx.x >> 5) & 3;
    int lane = threadIdx.x & 31;
    int ln = lane & 15, lh = lane >> 4;

    int qblk = wid & (N_ / 16 - 1);
    int h    = (wid >> 7) & (H_ - 1);
    int b    = wid >> 11;
    int q0   = qblk * 16;

    const bf16_t* qrow = Q + ((size_t)(b * N_ + q0 + ln)) * C_ + h * D_;
    v16bf qa0 = load_frag(qrow + lh * 8,      qrow + 16 + lh * 8);
    v16bf qa1 = load_frag(qrow + 32 + lh * 8, qrow + 48 + lh * 8);

    v8f ot[4] = {};
    v8f mrow, lrow;
#pragma unroll
    for (int r = 0; r < 8; ++r) { mrow[r] = -3.0e38f; lrow[r] = 0.0f; }

    int jend = (q0 + 15) >> 5;
    for (int jb = 0; jb <= jend; ++jb) {
        const bf16_t* kr0 = Kb + ((size_t)(b * N_ + jb * 32 + ln)) * C_ + h * D_;
        const bf16_t* kr1 = Kb + ((size_t)(b * N_ + jb * 32 + 16 + ln)) * C_ + h * D_;
        v16bf kb00 = load_frag(kr0 + lh * 8,      kr0 + 16 + lh * 8);
        v16bf kb01 = load_frag(kr0 + 32 + lh * 8, kr0 + 48 + lh * 8);
        v16bf kb10 = load_frag(kr1 + lh * 8,      kr1 + 16 + lh * 8);
        v16bf kb11 = load_frag(kr1 + 32 + lh * 8, kr1 + 48 + lh * 8);

        v8f s0 = {}, s1 = {};
        s0 = wmma_bf16(qa0, kb00, s0);
        s0 = wmma_bf16(qa1, kb01, s0);
        s1 = wmma_bf16(qa0, kb10, s1);
        s1 = wmma_bf16(qa1, kb11, s1);

        if (jb * 32 + 31 > q0) {              // diagonal block: causal mask
#pragma unroll
            for (int r = 0; r < 8; ++r) {
                int qg = q0 + r + 8 * lh;
                if (jb * 32 + ln > qg)       s0[r] = -3.0e38f;
                if (jb * 32 + 16 + ln > qg)  s1[r] = -3.0e38f;
            }
        }

#pragma unroll
        for (int r = 0; r < 8; ++r) {
            float rm = fmaxf(s0[r], s1[r]);
            rm = fmaxf(rm, __shfl_xor(rm, 1, 32));
            rm = fmaxf(rm, __shfl_xor(rm, 2, 32));
            rm = fmaxf(rm, __shfl_xor(rm, 4, 32));
            rm = fmaxf(rm, __shfl_xor(rm, 8, 32));
            float mn    = fmaxf(mrow[r], rm);
            float alpha = __expf(mrow[r] - mn);
            float p0 = __expf(s0[r] - mn);
            float p1 = __expf(s1[r] - mn);
            s0[r] = p0; s1[r] = p1;
            float rs = p0 + p1;
            rs += __shfl_xor(rs, 1, 32);
            rs += __shfl_xor(rs, 2, 32);
            rs += __shfl_xor(rs, 4, 32);
            rs += __shfl_xor(rs, 8, 32);
            lrow[r] = lrow[r] * alpha + rs;
            mrow[r] = mn;
#pragma unroll
            for (int t = 0; t < 4; ++t) ot[t][r] *= alpha;
        }

        // C-layout -> A-layout for P via per-wave LDS tile
        bf16_t* pl = &lds[wloc][0];
#pragma unroll
        for (int r = 0; r < 8; ++r) {
            int m = r + 8 * lh;
            pl[m * 32 + ln]      = (bf16_t)s0[r];
            pl[m * 32 + 16 + ln] = (bf16_t)s1[r];
        }
        const bf16_t* prow = pl + ln * 32;
        v16bf pa = load_frag(prow + lh * 8, prow + 16 + lh * 8);

#pragma unroll
        for (int t = 0; t < 4; ++t) {
            const bf16_t* vrow = Vt + ((size_t)((b * H_ + h) * D_ + t * 16 + ln)) * N_ + jb * 32;
            v16bf vb = load_frag(vrow + lh * 8, vrow + 16 + lh * 8);
            ot[t] = wmma_bf16(pa, vb, ot[t]);
        }
    }

#pragma unroll
    for (int r = 0; r < 8; ++r) {
        float invl = 1.0f / lrow[r];
        int qr = q0 + r + 8 * lh;
#pragma unroll
        for (int t = 0; t < 4; ++t)
            Ob[((size_t)(b * N_ + qr)) * C_ + h * D_ + t * 16 + ln] = (bf16_t)(ot[t][r] * invl);
    }
}

// ---------------- host-side orchestration ----------------
extern "C" void kernel_launch(void* const* d_in, const int* in_sizes, int n_in,
                              void* d_out, int out_size, void* d_ws, size_t ws_size,
                              hipStream_t stream) {
    const float* hs   = (const float*)d_in[0];
    const float* cosb = (const float*)d_in[1];
    const float* sinb = (const float*)d_in[2];
    const float* Wq   = (const float*)d_in[3];
    const float* Wk   = (const float*)d_in[4];
    const float* Wv   = (const float*)d_in[5];
    const float* Wo   = (const float*)d_in[6];
    float* out = (float*)d_out;

    const size_t nHS = (size_t)B_ * N_ * C_;   // 4,194,304
    const size_t nW  = (size_t)C_ * C_;        // 1,048,576

    bf16_t* p = (bf16_t*)d_ws;
    bf16_t* hsB = p;  p += nHS;
    bf16_t* WqT = p;  p += nW;
    bf16_t* WkT = p;  p += nW;
    bf16_t* WvT = p;  p += nW;
    bf16_t* WoT = p;  p += nW;
    bf16_t* Qb  = p;  p += nHS;
    bf16_t* Kbb = p;  p += nHS;
    bf16_t* Vt  = p;  p += nHS;
    bf16_t* Obf = p;  p += nHS;    // also used as temp V before transpose

    // 1) convert inputs to bf16 (weights transposed)
    cvt_f32_bf16<<<(int)(nHS / 256), 256, 0, stream>>>(hs, hsB, (int)nHS);
    cvt_wT_bf16<<<(int)(nW / 256), 256, 0, stream>>>(Wq, WqT);
    cvt_wT_bf16<<<(int)(nW / 256), 256, 0, stream>>>(Wk, WkT);
    cvt_wT_bf16<<<(int)(nW / 256), 256, 0, stream>>>(Wv, WvT);
    cvt_wT_bf16<<<(int)(nW / 256), 256, 0, stream>>>(Wo, WoT);

    // 2) QKV projections (TDM-staged B panels): 512 blocks x 256 threads
    gemm_bf16_tdm<<<512, 256, 0, stream>>>(hsB, WqT, nullptr, Qb);
    gemm_bf16_tdm<<<512, 256, 0, stream>>>(hsB, WkT, nullptr, Kbb);
    gemm_bf16_tdm<<<512, 256, 0, stream>>>(hsB, WvT, nullptr, Obf);  // V temp

    // 3) RoPE (scale D^-0.5 folded into Q), V transpose
    const int nRope = B_ * N_ * H_ * 32;
    rope_bf16<<<nRope / 256, 256, 0, stream>>>(Qb,  cosb, sinb, 0.125f);
    rope_bf16<<<nRope / 256, 256, 0, stream>>>(Kbb, cosb, sinb, 1.0f);
    vT_bf16<<<(int)(nHS / 256), 256, 0, stream>>>(Obf, Vt);

    // 4) flash attention: B*H*(N/16)=4096 waves, 4 waves/block
    flash_attn<<<1024, 128, 0, stream>>>(Qb, Kbb, Vt, Obf);

    // 5) output projection -> f32 d_out
    gemm_bf16_tdm<<<512, 256, 0, stream>>>(Obf, WoT, out, nullptr);
}
